// VGRAMMemory_31310311587961
// MI455X (gfx1250) — 8-bit matrix operand order repaired and checked by reading, compiled, not run on hardware
//
#include <hip/hip_runtime.h>

typedef __attribute__((ext_vector_type(8))) int v8i;

namespace {
constexpr int C_N = 256;
constexpr int M_N = 1024;
constexpr int P_N = 512;
constexpr int D_N = 16;
constexpr int B_N = 64;
constexpr int CHUNK  = 128;
constexpr int NCHUNK = M_N / CHUNK;
constexpr int KSTEPS = P_N / 64;

constexpr int BITS_PITCH  = 520;
constexpr int PAT_PITCH   = 528;
constexpr int SCORE_PITCH = 132;
constexpr int V_PITCH     = 20;

constexpr int PAT_OFF    = B_N * BITS_PITCH;
constexpr int V_OFF      = PAT_OFF + CHUNK * PAT_PITCH;
constexpr int ASUM_OFF   = V_OFF + CHUNK * V_PITCH * 4;
constexpr int SMEM_BYTES = ASUM_OFF + CHUNK * 4;
}

__device__ __forceinline__ unsigned pack_pos4(float4 f) {
    const unsigned u0 = __float_as_uint(f.x);
    const unsigned u1 = __float_as_uint(f.y);
    const unsigned u2 = __float_as_uint(f.z);
    const unsigned u3 = __float_as_uint(f.w);
    const unsigned h01 = __builtin_amdgcn_perm(u1, u0, 0x00000703u);
    const unsigned h23 = __builtin_amdgcn_perm(u3, u2, 0x07030000u);
    const unsigned hb  = __builtin_amdgcn_perm(h23, h01, 0x07060100u);
    return (~hb >> 7) & 0x01010101u;
}

__global__ __launch_bounds__(256)
void vgram_wmma_kernel(const float* __restrict__ bits,
                       const float* __restrict__ plog,
                       const float* __restrict__ vlog,
                       float* __restrict__ out)
{
    __shared__ __align__(16) unsigned char smem[SMEM_BYTES];
    unsigned char* bitsLDS  = smem;
    unsigned char* patLDS   = smem + PAT_OFF;
    int*           scoreLDS = (int*)(smem + PAT_OFF);
    float*         vLDS     = (float*)(smem + V_OFF);
    int*           aSum     = (int*)(smem + ASUM_OFF);

    const int c    = blockIdx.x;
    const int tid  = threadIdx.x;
    const int wave = tid >> 5;
    const int lane = tid & 31;

    {
        const int b = tid >> 2;
        const int q = tid & 3;
        const float* src = bits + (size_t)b * (C_N * P_N) + (size_t)c * P_N + q * 128;
        unsigned char* dst = bitsLDS + b * BITS_PITCH + q * 128;
        #pragma unroll 8
        for (int i = 0; i < 32; ++i) {
            const float4 f = ((const float4*)src)[i];
            const unsigned int pk = (f.x > 0.5f ? 1u : 0u)
                                  | ((f.y > 0.5f ? 1u : 0u) << 8)
                                  | ((f.z > 0.5f ? 1u : 0u) << 16)
                                  | ((f.w > 0.5f ? 1u : 0u) << 24);
            ((unsigned int*)dst)[i] = pk;
        }
    }

    const int tb   = wave >> 1;
    const int tm0  = (wave & 1) * 4;
    const int l15  = lane & 15;
    const int lhi  = lane >> 4;
    const unsigned char* abase = bitsLDS + (tb * 16 + l15) * BITS_PITCH + lhi * 16;
    const unsigned char* bbase0 = patLDS + (tm0 * 16 + l15) * PAT_PITCH + lhi * 16;

    const int ab = tid & 63;
    const int g  = tid >> 6;
    float dmin = 3.4e38f, S = 0.0f;
    float ac0 = 0.0f, ac1 = 0.0f, ac2 = 0.0f, ac3 = 0.0f;
    int imin = 0;

    for (int ch = 0; ch < NCHUNK; ++ch) {
        const int mBase = ch * CHUNK;

        if (tid < CHUNK) aSum[tid] = 0;
        __syncthreads();

        {
            const int m = tid >> 1;
            const int h = tid & 1;
            const float* src = plog + ((size_t)c * M_N + (mBase + m)) * P_N + h * 256;
            unsigned char* dst = patLDS + m * PAT_PITCH + h * 256;
            if (ch + 1 < NCHUNK) __builtin_prefetch(src + CHUNK * P_N, 0, 0);
            int s = 0;
            #pragma unroll 8
            for (int i = 0; i < 64; ++i) {
                const unsigned pk = pack_pos4(((const float4*)src)[i]);
                ((unsigned int*)dst)[i] = pk;
                s += __popc(pk);
            }
            atomicAdd(&aSum[m], s);
        }
        for (int idx = tid; idx < CHUNK * D_N; idx += 256) {
            const int m = idx >> 4, d = idx & 15;
            const float x = vlog[((size_t)c * M_N + (mBase + m)) * D_N + d];
            vLDS[m * V_PITCH + d] = x > 0.0f ? 1.0f : 0.0f;
        }
        __syncthreads();

        v8i a0, a1, a2, a3;
        #pragma unroll
        for (int r = 0; r < 8; ++r) { a0[r] = 0; a1[r] = 0; a2[r] = 0; a3[r] = 0; }

        #pragma unroll
        for (int k = 0; k < KSTEPS; ++k) {
            const int k0 = k * 64;
            v8i afrag;
            {
                unsigned long long* av = (unsigned long long*)&afrag;
                av[0] = *(const unsigned long long*)(abase + k0);
                av[1] = *(const unsigned long long*)(abase + k0 + 8);
                av[2] = *(const unsigned long long*)(abase + k0 + 32);
                av[3] = *(const unsigned long long*)(abase + k0 + 40);
            }
            v8i bfrag;
            uint4* bv = (uint4*)&bfrag;

            bv[0] = *(const uint4*)(bbase0 + k0);
            bv[1] = *(const uint4*)(bbase0 + k0 + 32);
            a0 = __builtin_amdgcn_wmma_i32_16x16x64_iu8(false, afrag, false, bfrag, a0, false, false);

            bv[0] = *(const uint4*)(bbase0 + 16 * PAT_PITCH + k0);
            bv[1] = *(const uint4*)(bbase0 + 16 * PAT_PITCH + k0 + 32);
            a1 = __builtin_amdgcn_wmma_i32_16x16x64_iu8(false, afrag, false, bfrag, a1, false, false);

            bv[0] = *(const uint4*)(bbase0 + 32 * PAT_PITCH + k0);
            bv[1] = *(const uint4*)(bbase0 + 32 * PAT_PITCH + k0 + 32);
            a2 = __builtin_amdgcn_wmma_i32_16x16x64_iu8(false, afrag, false, bfrag, a2, false, false);

            bv[0] = *(const uint4*)(bbase0 + 48 * PAT_PITCH + k0);
            bv[1] = *(const uint4*)(bbase0 + 48 * PAT_PITCH + k0 + 32);
            a3 = __builtin_amdgcn_wmma_i32_16x16x64_iu8(false, afrag, false, bfrag, a3, false, false);
        }
        __syncthreads();

        {
            const int srow0 = tb * 16 + lhi * 8;
            const int ml0   = tm0 * 16 + l15;
            #pragma unroll
            for (int r = 0; r < 8; ++r) {
                int* srow = scoreLDS + (srow0 + r) * SCORE_PITCH + ml0;
                srow[0]  = a0[r];
                srow[16] = a1[r];
                srow[32] = a2[r];
                srow[48] = a3[r];
            }
        }
        __syncthreads();

        #pragma unroll 4
        for (int ml = 0; ml < CHUNK; ++ml) {
            const int cross = scoreLDS[ab * SCORE_PITCH + ml];
            const float d = (float)(aSum[ml] - 2 * cross);
            const float4 vv = *(const float4*)(vLDS + ml * V_PITCH + g * 4);
            if (d < dmin) {
                const float t = __expf(d - dmin);
                S   = S   * t + 1.0f;
                ac0 = ac0 * t + vv.x;
                ac1 = ac1 * t + vv.y;
                ac2 = ac2 * t + vv.z;
                ac3 = ac3 * t + vv.w;
                dmin = d;
                imin = mBase + ml;
            } else {
                const float w = __expf(dmin - d);
                S   += w;
                ac0 += w * vv.x;
                ac1 += w * vv.y;
                ac2 += w * vv.z;
                ac3 += w * vv.w;
            }
        }
        __syncthreads();
    }

    {
        const float invS = 1.0f / S;
        const float* vl = vlog + ((size_t)c * M_N + imin) * D_N + g * 4;
        float4 o;
        o.x = ac0 * invS; o.y = ac1 * invS; o.z = ac2 * invS; o.w = ac3 * invS;
        const float h0 = vl[0] > 0.0f ? 1.0f : 0.0f;
        const float h1 = vl[1] > 0.0f ? 1.0f : 0.0f;
        const float h2 = vl[2] > 0.0f ? 1.0f : 0.0f;
        const float h3 = vl[3] > 0.0f ? 1.0f : 0.0f;
        o.x = o.x + (h0 - o.x);
        o.y = o.y + (h1 - o.y);
        o.z = o.z + (h2 - o.z);
        o.w = o.w + (h3 - o.w);
        float* op = out + ((size_t)ab * C_N + c) * D_N + g * 4;
        *(float4*)op = o;
    }
}

extern "C" void kernel_launch(void* const* d_in, const int* in_sizes, int n_in,
                              void* d_out, int out_size, void* d_ws, size_t ws_size,
                              hipStream_t stream) {
    (void)in_sizes; (void)n_in; (void)out_size; (void)d_ws; (void)ws_size;
    const float* bits = (const float*)d_in[0];
    const float* plog = (const float*)d_in[1];
    const float* vlog = (const float*)d_in[2];
    float* out = (float*)d_out;
    dim3 grid(C_N), block(256);
    vgram_wmma_kernel<<<grid, block, 0, stream>>>(bits, plog, vlog, out);
}
